// KmeansObjectTokens_16767552323839
// MI455X (gfx1250) — matrix operand order repaired and checked by reading, not compiled, not run on hardware
//
#include <hip/hip_runtime.h>
#include <hip/hip_bf16.h>

// ---------------------------------------------------------------------------
// Batched spherical K-means on MI455X (gfx1250).
//   B=32, N=4096, C=256, K=64, 20 iterations.
// x (bf16, 64MB) is L2-resident on MI455X (192MB L2) -> every iteration
// streams from L2. cos similarities computed with v_wmma_f32_16x16x32_bf16.
// ---------------------------------------------------------------------------

#define KB_ 32
#define KN_ 4096
#define KC_ 256
#define KK_ 64
#define KITERS_ 20
#define TILE_N_ 512            // rows per workgroup (8 waves * 4 subtiles * 16)
#define CSTRIDE_ 264           // padded bf16 row stride for LDS centroids (bank-conflict free)

typedef __bf16 bf16_t;
typedef __attribute__((ext_vector_type(4)))  bf16_t v4bf;
typedef __attribute__((ext_vector_type(8)))  bf16_t v8bf;
typedef __attribute__((ext_vector_type(16))) bf16_t v16bf;
typedef __attribute__((ext_vector_type(8)))  float  v8f;

// ---------------------------------------------------------------------------
// 1) fp32 -> bf16 conversion of the whole feature tensor (runs once)
// ---------------------------------------------------------------------------
__global__ void kconvert(const float* __restrict__ x, bf16_t* __restrict__ xbf, int n4) {
    int i = blockIdx.x * blockDim.x + threadIdx.x;
    if (i >= n4) return;
    const float4 v = reinterpret_cast<const float4*>(x)[i];
    v4bf o;
    o[0] = (bf16_t)v.x; o[1] = (bf16_t)v.y; o[2] = (bf16_t)v.z; o[3] = (bf16_t)v.w;
    reinterpret_cast<v4bf*>(xbf)[i] = o;
}

// ---------------------------------------------------------------------------
// 2) zero the accumulators (runs once; finalize re-zeros each iteration)
// ---------------------------------------------------------------------------
__global__ void kzero(float* __restrict__ p, int n) {
    int i = blockIdx.x * blockDim.x + threadIdx.x;
    if (i < n) p[i] = 0.0f;
}

// ---------------------------------------------------------------------------
// 3) gather initial centroids: fp32 into d_out (live centroid buffer) + bf16
// ---------------------------------------------------------------------------
__global__ void kinit_cent(const float* __restrict__ x, const int* __restrict__ init_idx,
                           float* __restrict__ cent, bf16_t* __restrict__ cbf) {
    const int blk = blockIdx.x;          // b*K + k
    const int c   = threadIdx.x;         // 0..C-1
    const int b   = blk / KK_;
    const int row = init_idx[blk];
    const float v = x[((size_t)b * KN_ + row) * KC_ + c];
    cent[(size_t)blk * KC_ + c] = v;
    cbf [(size_t)blk * KC_ + c] = (bf16_t)v;
}

// ---------------------------------------------------------------------------
// 4) assignment + partial accumulation.
//    grid = (N/TILE_N, B), block = 256 (8 waves). Dynamic LDS:
//      [0, 33792)           centroids bf16, row stride CSTRIDE_
//      [33792, 33792+65536) fp32 partial sums  K*C
//      [+65536, +65792)     fp32 partial counts K
// ---------------------------------------------------------------------------
__global__ void kassign(const bf16_t* __restrict__ xbf, const bf16_t* __restrict__ cbf,
                        float* __restrict__ gsum, float* __restrict__ gcnt) {
    extern __shared__ char smem[];
    bf16_t* sc   = reinterpret_cast<bf16_t*>(smem);                       // K * CSTRIDE_
    float*  sp   = reinterpret_cast<float*>(smem + KK_ * CSTRIDE_ * 2);   // K * C
    float*  scnt = sp + KK_ * KC_;                                        // K

    const int b    = blockIdx.y;
    const int n0   = blockIdx.x * TILE_N_;
    const int tid  = threadIdx.x;
    const int wave = tid >> 5;
    const int lane = tid & 31;
    const int half = lane >> 4;   // 0: lanes 0-15, 1: lanes 16-31
    const int mrow = lane & 15;

    // ---- stage centroids into LDS (padded rows), zero partials -------------
    for (int idx = tid; idx < (KK_ * KC_) / 8; idx += 256) {
        const int k  = idx >> 5;            // C/8 = 32 chunks per centroid row
        const int cc = (idx & 31) * 8;
        *reinterpret_cast<v8bf*>(sc + k * CSTRIDE_ + cc) =
            *reinterpret_cast<const v8bf*>(cbf + ((size_t)b * KK_ + k) * KC_ + cc);
    }
    for (int i = tid; i < KK_ * KC_; i += 256) sp[i] = 0.0f;
    if (tid < KK_) scnt[tid] = 0.0f;
    __syncthreads();

    // ---- per-wave: 4 subtiles of 16 rows each ------------------------------
    for (int s = 0; s < 4; ++s) {
        const int rowBase = n0 + wave * 64 + s * 16;
        v8f acc[4];
        #pragma unroll
        for (int nt = 0; nt < 4; ++nt) {
            v8f z = {0.f, 0.f, 0.f, 0.f, 0.f, 0.f, 0.f, 0.f};
            acc[nt] = z;
        }

        const bf16_t* xrow = xbf + ((size_t)b * KN_ + rowBase + mrow) * KC_;
        __builtin_prefetch(xrow, 0, 0);

        #pragma unroll
        for (int ks = 0; ks < 8; ++ks) {
            const int kb = ks * 32;
            // A fragment: 16-bit A 16x32 layout — lane<16: K {0..7,16..23},
            // lane>=16: K {8..15,24..31} (two 16B chunks).
            v8bf a0 = *reinterpret_cast<const v8bf*>(xrow + kb + half * 8);
            v8bf a1 = *reinterpret_cast<const v8bf*>(xrow + kb + 16 + half * 8);
            v16bf a = __builtin_shufflevector(a0, a1, 0,1,2,3,4,5,6,7,8,9,10,11,12,13,14,15);

            #pragma unroll
            for (int nt = 0; nt < 4; ++nt) {
                // B fragment: column n = nt*16 + (lane&15); same K order as A
                // (lane<16: K {0..7,16..23}, lane>=16: K {8..15,24..31}).
                const bf16_t* crow = sc + (nt * 16 + mrow) * CSTRIDE_ + kb + half * 8;
                v8bf b0 = *reinterpret_cast<const v8bf*>(crow);
                v8bf b1 = *reinterpret_cast<const v8bf*>(crow + 16);
                v16bf bb = __builtin_shufflevector(b0, b1, 0,1,2,3,4,5,6,7,8,9,10,11,12,13,14,15);
                acc[nt] = __builtin_amdgcn_wmma_f32_16x16x32_bf16(
                    false, a, false, bb, (short)0, acc[nt], false, false);
            }
        }

        // ---- argmax over K=64 per row (cross-lane butterflies, wave32) -----
        // C/D layout: VGPR r holds row M=r (lanes 0-15) / M=r+8 (lanes 16-31),
        // column N = lane&15 (+16*nt across accumulators).
        int bkLo[8], bkHi[8];
        #pragma unroll
        for (int r = 0; r < 8; ++r) {
            float bv = acc[0][r];
            int   bi = mrow;
            #pragma unroll
            for (int nt = 1; nt < 4; ++nt) {
                float v = acc[nt][r];
                int  ci = nt * 16 + mrow;
                if (v > bv) { bv = v; bi = ci; }
            }
            #pragma unroll
            for (int mask = 1; mask <= 8; mask <<= 1) {
                float ov = __shfl_xor(bv, mask, 16);
                int   oi = __shfl_xor(bi, mask, 16);
                if (ov > bv || (ov == bv && oi < bi)) { bv = ov; bi = oi; }
            }
            bkLo[r] = __shfl(bi, 0, 32);   // row rowBase + r
            bkHi[r] = __shfl(bi, 16, 32);  // row rowBase + r + 8
        }

        // ---- accumulate rows into LDS partial sums (ds_add_f32) ------------
        const int c0 = lane * 8;
        #pragma unroll
        for (int m = 0; m < 16; ++m) {
            const int kk = (m < 8) ? bkLo[m] : bkHi[m - 8];
            const bf16_t* xr = xbf + ((size_t)b * KN_ + rowBase + m) * KC_ + c0;
            v8bf xv = *reinterpret_cast<const v8bf*>(xr);
            #pragma unroll
            for (int i = 0; i < 8; ++i)
                atomicAdd(&sp[kk * KC_ + c0 + i], (float)xv[i]);
            if (lane == 0) atomicAdd(&scnt[kk], 1.0f);
        }
    }
    __syncthreads();

    // ---- flush workgroup partials to global accumulators -------------------
    for (int i = tid; i < KK_ * KC_; i += 256)
        atomicAdd(&gsum[(size_t)b * KK_ * KC_ + i], sp[i]);
    if (tid < KK_)
        atomicAdd(&gcnt[b * KK_ + tid], scnt[tid]);
}

// ---------------------------------------------------------------------------
// 5) finalize: mean (count clamped to 1), L2-normalize, write fp32 (d_out)
//    + bf16 copy, reset accumulators. grid = B*K, block = C = 256.
// ---------------------------------------------------------------------------
__global__ void kfinalize(float* __restrict__ cent, bf16_t* __restrict__ cbf,
                          float* __restrict__ gsum, float* __restrict__ gcnt) {
    __shared__ float red[8];
    const int blk = blockIdx.x;      // b*K + k
    const int c   = threadIdx.x;     // 0..C-1
    const int wv  = c >> 5;
    const int ln  = c & 31;

    const float s    = gsum[(size_t)blk * KC_ + c];
    const float cnt  = gcnt[blk];
    const float mean = s / fmaxf(cnt, 1.0f);

    float sq = mean * mean;
    #pragma unroll
    for (int off = 16; off >= 1; off >>= 1)
        sq += __shfl_down(sq, off, 32);
    if (ln == 0) red[wv] = sq;
    __syncthreads();
    if (c == 0) {
        float tot = 0.f;
        #pragma unroll
        for (int w = 0; w < 8; ++w) tot += red[w];
        red[0] = (tot > 0.f) ? rsqrtf(tot) : 0.f;
    }
    __syncthreads();
    const float out = mean * red[0];

    cent[(size_t)blk * KC_ + c] = out;
    cbf [(size_t)blk * KC_ + c] = (bf16_t)out;

    // reset accumulators for the next iteration
    gsum[(size_t)blk * KC_ + c] = 0.0f;
    if (c == 0) gcnt[blk] = 0.0f;
}

// ---------------------------------------------------------------------------
extern "C" void kernel_launch(void* const* d_in, const int* in_sizes, int n_in,
                              void* d_out, int out_size, void* d_ws, size_t ws_size,
                              hipStream_t stream) {
    (void)in_sizes; (void)n_in; (void)out_size; (void)ws_size;

    const float* x        = (const float*)d_in[0];   // [B, N, C] fp32, L2-normalized
    const int*   init_idx = (const int*)d_in[2];     // [B, K]
    float*       cent     = (float*)d_out;           // [B, K, C] fp32 (live centroids)

    // workspace layout
    bf16_t* xbf  = (bf16_t*)d_ws;                          // B*N*C bf16 (64 MB)
    bf16_t* cbf  = xbf + (size_t)KB_ * KN_ * KC_;          // B*K*C bf16
    float*  gsum = (float*)(cbf + (size_t)KB_ * KK_ * KC_);// B*K*C fp32
    float*  gcnt = gsum + (size_t)KB_ * KK_ * KC_;         // B*K   fp32

    // 1) convert x to bf16 (once) — stays L2-resident for all 20 iterations
    {
        const int n4 = (KB_ * KN_ * KC_) / 4;
        kconvert<<<(n4 + 255) / 256, 256, 0, stream>>>(x, xbf, n4);
    }
    // 2) zero accumulators (once)
    {
        const int nz = KB_ * KK_ * KC_ + KB_ * KK_;
        kzero<<<(nz + 255) / 256, 256, 0, stream>>>(gsum, nz);
    }
    // 3) initial centroids
    kinit_cent<<<KB_ * KK_, KC_, 0, stream>>>(x, init_idx, cent, cbf);

    // 4) Lloyd iterations
    const size_t smemBytes = (size_t)KK_ * CSTRIDE_ * 2   // bf16 centroids
                           + (size_t)KK_ * KC_ * 4        // fp32 partial sums
                           + (size_t)KK_ * 4;             // fp32 partial counts
    dim3 agrid(KN_ / TILE_N_, KB_);
    for (int it = 0; it < KITERS_; ++it) {
        kassign<<<agrid, 256, smemBytes, stream>>>(xbf, cbf, gsum, gcnt);
        kfinalize<<<KB_ * KK_, KC_, 0, stream>>>(cent, cbf, gsum, gcnt);
    }
}
